// TransformerClassifier_45913200394724
// MI455X (gfx1250) — compile-verified, matrix operand order checked by reading, not run on hardware
//
#include <hip/hip_runtime.h>
#include <math.h>

// ---------------------------------------------------------------------------
// Types for WMMA fragments (gfx1250, wave32)
// ---------------------------------------------------------------------------
typedef __attribute__((ext_vector_type(16))) _Float16 v16h;
typedef __attribute__((ext_vector_type(8)))  float    v8f;
typedef __attribute__((ext_vector_type(4)))  unsigned int v4u;

union FragH { v16h h; v4u u[2]; };
union Pack8H { _Float16 t[8]; v4u u; };

#define BM 128
#define BN 64
#define BK 32

// ---------------------------------------------------------------------------
// Generic WMMA GEMM:  C = alpha * (A @ B) + bias  [optional ReLU]
//   A : f16 global, row-major [M,K], leading dim lda (element units)
//   B : f32 global. BTRANS==0 -> B is [K,N] row-major (weights / V)
//                   BTRANS==1 -> B is [N,K] row-major (K-matrix for Q.K^T)
//   C : optional f32 out, Ch : optional f16 out (both ldc)
//   Batched over gridDim.z with per-operand offsets
//     off = (z / zdiv) * s?1 + (z % zdiv) * s?2   (element units)
// Block: 256 threads = 8 waves; 128x64 tile; wave -> 32x32 output via
// four v_wmma_f32_16x16x32_f16 per K-step (reuse-A/B hints chained).
// ---------------------------------------------------------------------------
template <int BTRANS, int RELU>
__global__ void __launch_bounds__(256)
gemm_wmma(const _Float16* __restrict__ A, const float* __restrict__ B,
          const float* __restrict__ bias, float* __restrict__ C,
          _Float16* __restrict__ Ch,
          int M, int N, int K, int lda, int ldb, int ldc,
          float alpha, int zdiv,
          long long sA1, long long sA2, long long sB1, long long sB2,
          long long sC1, long long sC2)
{
    __shared__ _Float16 As[BM][BK + 8];   // +8 halfs keeps 16B row alignment
    __shared__ _Float16 Bs[BN][BK + 8];   // stored as [n][k]

    const int tid  = threadIdx.x;
    const int lane = tid & 31;
    const int w    = tid >> 5;
    const int wm   = w & 3;     // 32-row M sub-tile 0..3
    const int wn   = w >> 2;    // 32-col N sub-tile 0..1

    const int z  = blockIdx.z;
    const int zq = z / zdiv, zr = z % zdiv;
    const long long offA = zq * sA1 + (long long)zr * sA2;
    const long long offB = zq * sB1 + (long long)zr * sB2;
    const long long offC = zq * sC1 + (long long)zr * sC2;

    const int m0 = blockIdx.y * BM;
    const int n0 = blockIdx.x * BN;

    // A loader: 128x32 halfs, 16 halfs/thread (two b128)
    const int amRow = tid >> 1;            // 0..127
    const int amCol = (tid & 1) * 16;      // 0 or 16
    // B loader (64x32): 8 elements/thread
    const int bRow = tid >> 2;             // 0..63
    const int bCol = (tid & 3) * 8;        // 0,8,16,24

    const int lhalf = lane & 15;
    const int kb    = (lane < 16) ? 0 : 8; // ISA 16-bit A/B fragment layout

    v8f acc00 = {}, acc01 = {}, acc10 = {}, acc11 = {};

    for (int k0 = 0; k0 < K; k0 += BK) {
        // ---- stage A tile (f16) ----
        {
            const _Float16* srcp = A + offA + (long long)(m0 + amRow) * lda + k0 + amCol;
            *(v4u*)&As[amRow][amCol]     = *(const v4u*)(srcp);
            *(v4u*)&As[amRow][amCol + 8] = *(const v4u*)(srcp + 8);
            if (k0 + BK < K)   // next-tile prefetch -> global_prefetch_b8
                __builtin_prefetch(srcp + BK, 0, 1);
        }
        // ---- stage B tile as Bs[n][k] (convert fp32 -> f16) ----
        if (BTRANS) {
            const float* src = B + offB + (long long)(n0 + bRow) * ldb + k0 + bCol;
            float4 f0 = *(const float4*)(src);
            float4 f1 = *(const float4*)(src + 4);
            Pack8H p;
            p.t[0] = (_Float16)f0.x; p.t[1] = (_Float16)f0.y;
            p.t[2] = (_Float16)f0.z; p.t[3] = (_Float16)f0.w;
            p.t[4] = (_Float16)f1.x; p.t[5] = (_Float16)f1.y;
            p.t[6] = (_Float16)f1.z; p.t[7] = (_Float16)f1.w;
            *(v4u*)&Bs[bRow][bCol] = p.u;
        } else {
            const int kk = tid & 31;
            const int ng = (tid >> 5) * 8;
            const float* src = B + offB + (long long)(k0 + kk) * ldb + n0 + ng;
            float4 f0 = *(const float4*)(src);
            float4 f1 = *(const float4*)(src + 4);
            float f[8] = {f0.x, f0.y, f0.z, f0.w, f1.x, f1.y, f1.z, f1.w};
#pragma unroll
            for (int j = 0; j < 8; ++j) Bs[ng + j][kk] = (_Float16)f[j];
        }
        __syncthreads();

        // ---- fragments per ISA layout: two contiguous 8-half runs ----
        FragH a0, a1, b0, b1;
        const int ar0 = wm * 32 + lhalf;
        a0.u[0] = *(const v4u*)&As[ar0][kb];
        a0.u[1] = *(const v4u*)&As[ar0][kb + 16];
        a1.u[0] = *(const v4u*)&As[ar0 + 16][kb];
        a1.u[1] = *(const v4u*)&As[ar0 + 16][kb + 16];
        const int bn0 = wn * 32 + lhalf;
        b0.u[0] = *(const v4u*)&Bs[bn0][kb];
        b0.u[1] = *(const v4u*)&Bs[bn0][kb + 16];
        b1.u[0] = *(const v4u*)&Bs[bn0 + 16][kb];
        b1.u[1] = *(const v4u*)&Bs[bn0 + 16][kb + 16];

        // Chain so each reuse hint's NEXT wmma reuses the cached operand:
        // (a0,b0)[RA] -> (a0,b1)[RB] -> (a1,b1)[RA] -> (a1,b0)
        acc00 = __builtin_amdgcn_wmma_f32_16x16x32_f16(false, a0.h, false, b0.h,
                                                       (short)0, acc00, true,  false);
        acc01 = __builtin_amdgcn_wmma_f32_16x16x32_f16(false, a0.h, false, b1.h,
                                                       (short)0, acc01, false, true);
        acc11 = __builtin_amdgcn_wmma_f32_16x16x32_f16(false, a1.h, false, b1.h,
                                                       (short)0, acc11, true,  false);
        acc10 = __builtin_amdgcn_wmma_f32_16x16x32_f16(false, a1.h, false, b0.h,
                                                       (short)0, acc10, false, false);
        __syncthreads();
    }

    // ---- epilogue: 16x16 f32 D layout (VGPR r: rows r / r+8) ----
    const int rowBase = m0 + wm * 32 + ((lane >> 4) << 3);
    const int colBase = n0 + wn * 32 + lhalf;
    v8f accs[2][2] = {{acc00, acc01}, {acc10, acc11}};
#pragma unroll
    for (int ni = 0; ni < 2; ++ni) {
        const int col = colBase + ni * 16;
        const float bz = bias ? bias[col] : 0.0f;
#pragma unroll
        for (int mi = 0; mi < 2; ++mi) {
#pragma unroll
            for (int r = 0; r < 8; ++r) {
                const int row = rowBase + mi * 16 + r;
                float v = accs[mi][ni][r] * alpha + bz;
                if (RELU) v = fmaxf(v, 0.0f);
                const long long i = offC + (long long)row * ldc + col;
                if (C)  C[i]  = v;
                if (Ch) Ch[i] = (_Float16)v;
            }
        }
    }
}

// ---------------------------------------------------------------------------
// Embedding * sqrt(D) + sinusoidal positional encoding -> x (f32) + xh (f16)
// ---------------------------------------------------------------------------
__global__ void __launch_bounds__(256)
embed_pe_kernel(const int* __restrict__ text, const float* __restrict__ emb,
                float* __restrict__ x, _Float16* __restrict__ xh, int seq, int dim)
{
    const int row = blockIdx.x;           // b*seq + s
    const int s   = row % seq;
    const int tok = text[row];
    const float scale = sqrtf((float)dim);
    for (int d = threadIdx.x; d < dim; d += 256) {
        float v = emb[(long long)tok * dim + d] * scale;
        const int i2 = d & ~1;
        const float freq = expf((float)i2 * (-logf(10000.0f) / (float)dim));
        v += (d & 1) ? cosf((float)s * freq) : sinf((float)s * freq);
        x[(long long)row * dim + d]  = v;
        xh[(long long)row * dim + d] = (_Float16)v;
    }
}

// ---------------------------------------------------------------------------
// Row softmax with mask: probs (f16) = softmax(mask ? scores : -1e9). S = 512.
// ---------------------------------------------------------------------------
__global__ void __launch_bounds__(256)
softmax_mask_kernel(const float* __restrict__ scores, _Float16* __restrict__ probs,
                    const int* __restrict__ mask, int seq, int nh)
{
    __shared__ float red[256];
    const int row = blockIdx.x;           // z*seq + q,  z = b*nh + h
    const int z   = row / seq;
    const int b   = z / nh;
    const int tid = threadIdx.x;
    const float* src = scores + (long long)row * seq;

    float v0 = src[tid];
    float v1 = src[tid + 256];
    if (mask[b * seq + tid] == 0)       v0 = -1e9f;
    if (mask[b * seq + tid + 256] == 0) v1 = -1e9f;

    red[tid] = fmaxf(v0, v1);
    __syncthreads();
    for (int s = 128; s > 0; s >>= 1) { if (tid < s) red[tid] = fmaxf(red[tid], red[tid + s]); __syncthreads(); }
    const float m = red[0];
    __syncthreads();

    const float e0 = __expf(v0 - m), e1 = __expf(v1 - m);
    red[tid] = e0 + e1;
    __syncthreads();
    for (int s = 128; s > 0; s >>= 1) { if (tid < s) red[tid] += red[tid + s]; __syncthreads(); }
    const float inv = 1.0f / red[0];

    probs[(long long)row * seq + tid]       = (_Float16)(e0 * inv);
    probs[(long long)row * seq + tid + 256] = (_Float16)(e1 * inv);
}

// ---------------------------------------------------------------------------
// x = LayerNorm(x + resid) * g + b   (resid may be null); also emit f16 copy.
// D = 768 = 3 * 256.
// ---------------------------------------------------------------------------
__global__ void __launch_bounds__(256)
resln_kernel(float* __restrict__ x, const float* __restrict__ resid,
             const float* __restrict__ g, const float* __restrict__ bb,
             _Float16* __restrict__ xh, int dim)
{
    __shared__ float red[256];
    const int row = blockIdx.x;
    const int tid = threadIdx.x;
    const long long base = (long long)row * dim;

    float t[3];
#pragma unroll
    for (int i = 0; i < 3; ++i) {
        const int d = tid + i * 256;
        t[i] = x[base + d] + (resid ? resid[base + d] : 0.0f);
    }
    red[tid] = t[0] + t[1] + t[2];
    __syncthreads();
    for (int s = 128; s > 0; s >>= 1) { if (tid < s) red[tid] += red[tid + s]; __syncthreads(); }
    const float mu = red[0] / (float)dim;
    __syncthreads();

    float sq = 0.0f;
#pragma unroll
    for (int i = 0; i < 3; ++i) { const float d = t[i] - mu; sq += d * d; }
    red[tid] = sq;
    __syncthreads();
    for (int s = 128; s > 0; s >>= 1) { if (tid < s) red[tid] += red[tid + s]; __syncthreads(); }
    const float rinv = rsqrtf(red[0] / (float)dim + 1e-5f);

#pragma unroll
    for (int i = 0; i < 3; ++i) {
        const int d = tid + i * 256;
        const float y = (t[i] - mu) * rinv * g[d] + bb[d];
        x[base + d]  = y;
        xh[base + d] = (_Float16)y;
    }
}

// ---------------------------------------------------------------------------
// Mean-pool over sequence, FC (768 -> C), log_softmax. One block per batch.
// ---------------------------------------------------------------------------
__global__ void __launch_bounds__(256)
head_kernel(const float* __restrict__ x, const float* __restrict__ Wfc,
            const float* __restrict__ bfc, float* __restrict__ out,
            int seq, int dim, int ncls)
{
    __shared__ float pooled[768];
    __shared__ float red[256];
    __shared__ float lg[8];
    const int b = blockIdx.x;
    const int tid = threadIdx.x;

    for (int d = tid; d < dim; d += 256) {
        float s = 0.0f;
        for (int t = 0; t < seq; ++t)
            s += x[(long long)(b * seq + t) * dim + d];
        pooled[d] = s / (float)seq;
    }
    __syncthreads();

    for (int c = 0; c < ncls; ++c) {
        float p = 0.0f;
        for (int d = tid; d < dim; d += 256) p += pooled[d] * Wfc[d * ncls + c];
        red[tid] = p;
        __syncthreads();
        for (int s = 128; s > 0; s >>= 1) { if (tid < s) red[tid] += red[tid + s]; __syncthreads(); }
        if (tid == 0) lg[c] = red[0] + bfc[c];
        __syncthreads();
    }
    if (tid == 0) {
        float m = lg[0];
        for (int c = 1; c < ncls; ++c) m = fmaxf(m, lg[c]);
        float s = 0.0f;
        for (int c = 0; c < ncls; ++c) s += __expf(lg[c] - m);
        const float lse = m + __logf(s);
        for (int c = 0; c < ncls; ++c) out[b * ncls + c] = lg[c] - lse;
    }
}

// ---------------------------------------------------------------------------
// Host orchestration
// ---------------------------------------------------------------------------
extern "C" void kernel_launch(void* const* d_in, const int* in_sizes, int n_in,
                              void* d_out, int out_size, void* d_ws, size_t ws_size,
                              hipStream_t stream)
{
    (void)in_sizes; (void)n_in; (void)out_size; (void)ws_size;

    constexpr int L = 4, D = 768, H = 12, FF = 3072, B = 2, S = 512, CLS = 2;
    constexpr int M = B * S;          // 1024 token rows
    constexpr int HD = D / H;         // 64

    const int*   text  = (const int*)  d_in[0];
    const int*   mask  = (const int*)  d_in[1];
    const float* emb   = (const float*)d_in[2];
    const float* Wq    = (const float*)d_in[3];
    const float* bq    = (const float*)d_in[4];
    const float* Wk    = (const float*)d_in[5];
    const float* bk    = (const float*)d_in[6];
    const float* Wv    = (const float*)d_in[7];
    const float* bv    = (const float*)d_in[8];
    const float* Wo    = (const float*)d_in[9];
    const float* bo    = (const float*)d_in[10];
    const float* W1    = (const float*)d_in[11];
    const float* b1    = (const float*)d_in[12];
    const float* W2    = (const float*)d_in[13];
    const float* b2    = (const float*)d_in[14];
    const float* ln1_g = (const float*)d_in[15];
    const float* ln1_b = (const float*)d_in[16];
    const float* ln2_g = (const float*)d_in[17];
    const float* ln2_b = (const float*)d_in[18];
    const float* lnf_g = (const float*)d_in[19];
    const float* lnf_b = (const float*)d_in[20];
    const float* Wfc   = (const float*)d_in[21];
    const float* bfc   = (const float*)d_in[22];

    // ---- workspace carve-out (all buffers fully written before read) ----
    char* ws = (char*)d_ws;
    size_t off = 0;
    auto alloc = [&](size_t bytes) -> void* {
        void* p = ws + off;
        off += (bytes + 255) & ~(size_t)255;
        return p;
    };
    float*     x    = (float*)    alloc((size_t)M * D * 4);
    _Float16*  xh   = (_Float16*) alloc((size_t)M * D * 2);
    _Float16*  qh   = (_Float16*) alloc((size_t)M * D * 2);
    float*     kf   = (float*)    alloc((size_t)M * D * 4);
    float*     vf   = (float*)    alloc((size_t)M * D * 4);
    float*     sc   = (float*)    alloc((size_t)B * H * S * S * 4);
    _Float16*  pr   = (_Float16*) alloc((size_t)B * H * S * S * 2);
    _Float16*  aoh  = (_Float16*) alloc((size_t)M * D * 2);
    float*     proj = (float*)    alloc((size_t)M * D * 4);
    _Float16*  h1h  = (_Float16*) alloc((size_t)M * FF * 2);

    auto gemm = [&](const _Float16* Ap, const float* Bp, const float* biasp,
                    float* Cp, _Float16* Chp,
                    int Mm, int Nn, int Kk, int lda, int ldb, int ldc,
                    float alpha, int bT, int relu,
                    int Z, int zdiv,
                    long long a1, long long a2, long long bs1, long long bs2,
                    long long c1, long long c2) {
        dim3 grid(Nn / BN, Mm / BM, Z);
        if (bT)
            gemm_wmma<1, 0><<<grid, 256, 0, stream>>>(Ap, Bp, biasp, Cp, Chp,
                Mm, Nn, Kk, lda, ldb, ldc, alpha, zdiv, a1, a2, bs1, bs2, c1, c2);
        else if (relu)
            gemm_wmma<0, 1><<<grid, 256, 0, stream>>>(Ap, Bp, biasp, Cp, Chp,
                Mm, Nn, Kk, lda, ldb, ldc, alpha, zdiv, a1, a2, bs1, bs2, c1, c2);
        else
            gemm_wmma<0, 0><<<grid, 256, 0, stream>>>(Ap, Bp, biasp, Cp, Chp,
                Mm, Nn, Kk, lda, ldb, ldc, alpha, zdiv, a1, a2, bs1, bs2, c1, c2);
    };

    embed_pe_kernel<<<M, 256, 0, stream>>>(text, emb, x, xh, S, D);

    const long long SD = (long long)S * D;
    const long long SS = (long long)S * S;

    for (int l = 0; l < L; ++l) {
        const float* wq = Wq + (size_t)l * D * D;  const float* bql = bq + l * D;
        const float* wk = Wk + (size_t)l * D * D;  const float* bkl = bk + l * D;
        const float* wv = Wv + (size_t)l * D * D;  const float* bvl = bv + l * D;
        const float* wo = Wo + (size_t)l * D * D;  const float* bol = bo + l * D;
        const float* w1 = W1 + (size_t)l * D * FF; const float* b1l = b1 + l * FF;
        const float* w2 = W2 + (size_t)l * FF * D; const float* b2l = b2 + l * D;
        const float* g1 = ln1_g + l * D; const float* be1 = ln1_b + l * D;
        const float* g2 = ln2_g + l * D; const float* be2 = ln2_b + l * D;

        // Q, K, V projections
        gemm(xh, wq, bql, nullptr, qh, M, D, D, D, D, D, 1.0f, 0, 0, 1, 1, 0, 0, 0, 0, 0, 0);
        gemm(xh, wk, bkl, kf, nullptr, M, D, D, D, D, D, 1.0f, 0, 0, 1, 1, 0, 0, 0, 0, 0, 0);
        gemm(xh, wv, bvl, vf, nullptr, M, D, D, D, D, D, 1.0f, 0, 0, 1, 1, 0, 0, 0, 0, 0, 0);

        // scores[z] = (Q_h @ K_h^T) / sqrt(64), batched over z = b*H + h
        gemm(qh, kf, nullptr, sc, nullptr, S, S, HD, D, D, S, 0.125f, 1, 0,
             B * H, H, SD, HD, SD, HD, (long long)H * SS, SS);

        softmax_mask_kernel<<<B * H * S, 256, 0, stream>>>(sc, pr, mask, S, H);

        // out[z] = probs @ V_h  (f16 result into head columns of aoh)
        gemm(pr, vf, nullptr, nullptr, aoh, S, HD, S, S, D, D, 1.0f, 0, 0,
             B * H, H, (long long)H * SS, SS, SD, HD, SD, HD);

        // output projection + residual LN1
        gemm(aoh, wo, bol, proj, nullptr, M, D, D, D, D, D, 1.0f, 0, 0, 1, 1, 0, 0, 0, 0, 0, 0);
        resln_kernel<<<M, 256, 0, stream>>>(x, proj, g1, be1, xh, D);

        // FFN
        gemm(xh, w1, b1l, nullptr, h1h, M, FF, D, D, FF, FF, 1.0f, 0, 1, 1, 1, 0, 0, 0, 0, 0, 0);
        gemm(h1h, w2, b2l, proj, nullptr, M, D, FF, FF, D, D, 1.0f, 0, 0, 1, 1, 0, 0, 0, 0, 0, 0);
        resln_kernel<<<M, 256, 0, stream>>>(x, proj, g2, be2, xh, D);
    }

    // final LayerNorm (no residual), then pool + classifier head
    resln_kernel<<<M, 256, 0, stream>>>(x, nullptr, lnf_g, lnf_b, xh, D);
    head_kernel<<<B, 256, 0, stream>>>(x, Wfc, bfc, (float*)d_out, S, D, CLS);
}
